// KendallDistance_68513318306381
// MI455X (gfx1250) — 8-bit matrix operand order repaired and checked by reading, compiled, not run on hardware
//
#include <hip/hip_runtime.h>

typedef __attribute__((ext_vector_type(8))) int v8i;

#define B_DIM 64
#define T_DIM 256
#define N_DIM 64
#define ST    76

__device__ __forceinline__ unsigned sgn_at(float d, int k) {
    unsigned pos = (d > 0.0f) ? (0x01u << (8 * k)) : 0u;
    return (d < 0.0f) ? (0xFFu << (8 * k)) : pos;
}

__device__ __forceinline__ int packsgn(float a, float b0, float b1, float b2, float b3) {
    return (int)(sgn_at(a - b0, 0) | sgn_at(a - b1, 1) |
                 sgn_at(a - b2, 2) | sgn_at(a - b3, 3));
}

__device__ __forceinline__ void accumulate_block(const float* stage, int l, int h,
                                                 int ab, int bb, v8i acc[4][4]) {
    float u[4][8], vv[4][8];
#pragma unroll
    for (int q = 0; q < 4; ++q) {
        const float4* pu = (const float4*)&stage[(q * 16 + l) * ST + ab * 8];
        float4 a0 = pu[0], a1 = pu[1];
        u[q][0] = a0.x; u[q][1] = a0.y; u[q][2] = a0.z; u[q][3] = a0.w;
        u[q][4] = a1.x; u[q][5] = a1.y; u[q][6] = a1.z; u[q][7] = a1.w;
        const float4* pv = (const float4*)&stage[(q * 16 + l) * ST + bb * 8];
        float4 b0 = pv[0], b1 = pv[1];
        vv[q][0] = b0.x; vv[q][1] = b0.y; vv[q][2] = b0.z; vv[q][3] = b0.w;
        vv[q][4] = b1.x; vv[q][5] = b1.y; vv[q][6] = b1.z; vv[q][7] = b1.w;
    }

    v8i A[4], Bf[4];
#pragma unroll
    for (int q = 0; q < 4; ++q) {
        const float v0 = vv[q][0], v1 = vv[q][1], v2 = vv[q][2], v3 = vv[q][3];
        const float v4 = vv[q][4], v5 = vv[q][5], v6 = vv[q][6], v7 = vv[q][7];
        const float uA0 = h ? u[q][1] : u[q][0];
        const float uA1 = h ? u[q][3] : u[q][2];
        const float uA2 = h ? u[q][5] : u[q][4];
        const float uA3 = h ? u[q][7] : u[q][6];
        const float uB0 = h ? u[q][1] : u[q][0];
        const float uB1 = h ? u[q][3] : u[q][2];
        const float uB2 = h ? u[q][5] : u[q][4];
        const float uB3 = h ? u[q][7] : u[q][6];

        A[q][0] = packsgn(uA0, v0, v1, v2, v3);
        A[q][1] = packsgn(uA0, v4, v5, v6, v7);
        A[q][2] = packsgn(uA1, v0, v1, v2, v3);
        A[q][3] = packsgn(uA1, v4, v5, v6, v7);
        A[q][4] = packsgn(uA2, v0, v1, v2, v3);
        A[q][5] = packsgn(uA2, v4, v5, v6, v7);
        A[q][6] = packsgn(uA3, v0, v1, v2, v3);
        A[q][7] = packsgn(uA3, v4, v5, v6, v7);

        Bf[q][0] = packsgn(uB0, v0, v1, v2, v3);
        Bf[q][1] = packsgn(uB0, v4, v5, v6, v7);
        Bf[q][2] = packsgn(uB1, v0, v1, v2, v3);
        Bf[q][3] = packsgn(uB1, v4, v5, v6, v7);
        Bf[q][4] = packsgn(uB2, v0, v1, v2, v3);
        Bf[q][5] = packsgn(uB2, v4, v5, v6, v7);
        Bf[q][6] = packsgn(uB3, v0, v1, v2, v3);
        Bf[q][7] = packsgn(uB3, v4, v5, v6, v7);
    }

#pragma unroll
    for (int qi = 0; qi < 4; ++qi)
#pragma unroll
        for (int qj = 0; qj < 4; ++qj)
            acc[qi][qj] = __builtin_amdgcn_wmma_i32_16x16x64_iu8(
                true, A[qi], true, Bf[qj], acc[qi][qj], false, false);
}

__device__ __forceinline__ void zero_acc(v8i acc[4][4]) {
#pragma unroll
    for (int qi = 0; qi < 4; ++qi)
#pragma unroll
        for (int qj = 0; qj < 4; ++qj) {
            v8i z = {0, 0, 0, 0, 0, 0, 0, 0};
            acc[qi][qj] = z;
        }
}

__device__ __forceinline__ void flush_acc(int* cpart, v8i acc[4][4],
                                          int l, int h, int scale) {
#pragma unroll
    for (int qi = 0; qi < 4; ++qi)
#pragma unroll
        for (int qj = 0; qj < 4; ++qj)
#pragma unroll
            for (int g = 0; g < 8; ++g)
                atomicAdd(&cpart[(qi * 16 + h * 8 + g) * N_DIM + qj * 16 + l],
                          scale * acc[qi][qj][g]);
}

__global__ __launch_bounds__(256)
void kendall_gram_kernel(const float* __restrict__ x, int* __restrict__ Cacc) {
    __shared__ __align__(16) float stage[N_DIM * ST];
    __shared__ int cpart[N_DIM * N_DIM];

    const int t   = blockIdx.x;
    const int tid = threadIdx.x;

    for (int k = tid; k < B_DIM * N_DIM; k += 256) {
        const int a = k >> 6;
        const int n = k & 63;
        stage[n * ST + a] = x[(a * T_DIM + t) * N_DIM + n];
    }
    for (int k = tid; k < N_DIM * N_DIM; k += 256) cpart[k] = 0;
    __syncthreads();

    const int lane = tid & 31;
    const int l    = lane & 15;
    const int h    = lane >> 4;
    const int wv   = tid >> 5;

    v8i acc[4][4];

    zero_acc(acc);
    accumulate_block(stage, l, h, wv, wv, acc);
    flush_acc(cpart, acc, l, h, 1);

    zero_acc(acc);
#pragma unroll 1
    for (int idx = wv; idx < 28; idx += 8) {
        int ab = 0, rem = idx;
        while (rem >= 7 - ab) { rem -= 7 - ab; ++ab; }
        const int bb = ab + 1 + rem;
        accumulate_block(stage, l, h, ab, bb, acc);
    }
    flush_acc(cpart, acc, l, h, 2);
    __syncthreads();

    for (int k = tid; k < N_DIM * N_DIM; k += 256)
        atomicAdd(&Cacc[k], cpart[k]);
}

__global__ void kendall_zero_kernel(int* __restrict__ C) {
    const int i = blockIdx.x * blockDim.x + threadIdx.x;
    if (i < N_DIM * N_DIM) C[i] = 0;
}

__global__ void kendall_finalize_kernel(const int* __restrict__ C,
                                        float* __restrict__ out) {
    const int idx = blockIdx.x * blockDim.x + threadIdx.x;
    if (idx >= B_DIM * N_DIM * N_DIM) return;
    const int ij = idx & (N_DIM * N_DIM - 1);
    const int i  = ij >> 6;
    const int j  = ij & 63;
    float d = 1.0f - (float)C[ij] / 2016.0f;
    if (i == j) d = 0.0f;
    out[idx] = d;
}

extern "C" void kernel_launch(void* const* d_in, const int* in_sizes, int n_in,
                              void* d_out, int out_size, void* d_ws, size_t ws_size,
                              hipStream_t stream) {
    const float* x = (const float*)d_in[0];
    int*   Cacc = (int*)d_ws;
    float* out  = (float*)d_out;

    kendall_zero_kernel<<<(N_DIM * N_DIM + 255) / 256, 256, 0, stream>>>(Cacc);
    kendall_gram_kernel<<<T_DIM, 256, 0, stream>>>(x, Cacc);
    kendall_finalize_kernel<<<(B_DIM * N_DIM * N_DIM + 255) / 256, 256, 0, stream>>>(Cacc, out);
}
